// BiSSMBlock_32693291057657
// MI455X (gfx1250) — matrix operand order repaired and checked by reading, compiled, not run on hardware
//
#include <hip/hip_runtime.h>
#include <hip/hip_bf16.h>
#include <cstdint>

// ---- problem constants (match reference) ----
#define DMODEL 1024
#define DSTATE 16
#define DCONV  4
#define DI     2048          // D_INNER
#define DTRANK 64
#define BATCH_ 2
#define LSEQ   1024
#define MTOK   (BATCH_*LSEQ) // 2048 tokens
#define XDBL_N (DTRANK + 2*DSTATE) // 96

typedef _Float16 half_t;
typedef __attribute__((ext_vector_type(16))) _Float16 v16h;
typedef __attribute__((ext_vector_type(8)))  _Float16 v8h;
typedef __attribute__((ext_vector_type(4)))  _Float16 v4h;
typedef __attribute__((ext_vector_type(8)))  float    v8f;
typedef __attribute__((ext_vector_type(4)))  float    v4f;
typedef __attribute__((ext_vector_type(4)))  int      v4i;

#define BM 128
#define BN 64
#define BK 32

// ---- CDNA5 async global->LDS copy (ASYNCcnt path), with safe fallback ----
#if __has_builtin(__builtin_amdgcn_global_load_async_to_lds_b128)
#define ASYNC_LDS 1
__device__ __forceinline__ void asyncCopy16(const void* g, void* l) {
  // param 1: LDS destination (non-const v4i* per toolchain signature);
  // param 2: global source; imm offset = 0, cpol = 0.
  __builtin_amdgcn_global_load_async_to_lds_b128((v4i*)l, (v4i*)g, 0, 0);
}
__device__ __forceinline__ void waitAsync() {
#if __has_builtin(__builtin_amdgcn_s_wait_asynccnt)
  __builtin_amdgcn_s_wait_asynccnt(0);
#else
  asm volatile("s_wait_asynccnt 0x0" ::: "memory");
#endif
}
#else
#define ASYNC_LDS 0
#endif

__device__ __forceinline__ float siluf(float x) { return x / (1.f + __expf(-x)); }

__device__ __forceinline__ int flipRow(int m) {
  int t = m & (LSEQ - 1);
  return (m - t) + (LSEQ - 1 - t);
}

// Epilog: E=0 bias add; E=1 softplus(bias add)
template<int E>
__device__ __forceinline__ float epilog(float acc, float b) {
  float v = acc + b;
  if (E == 1) v = (v > 20.f) ? v : __logf(1.f + __expf(v));
  return v;
}

// C(MxN) = epilog( prologA(MxK) @ Wh(NxK)^T + bias ), Wh pre-converted to f16.
// Prolog P: 0 = plain, 1 = time-flipped rows, 2 = A0*silu(A1), 3 = concat(A0 | flip(A1)) at K0.
// Relied on: M % BM == 0, K % BK == 0, K0 % BK == 0. Only N may be ragged (N=96 case).
// 256 threads = 8 wave32; wave grid 4(M) x 2(N); each wave: 32x32 via 4 WMMA tiles.
// A: register-pipelined global->LDS with f32->f16 cvt + fusion.
// B: double-buffered LDS tile fed by GLOBAL_LOAD_ASYNC_TO_LDS_B128 (or reg fallback).
template<int P, int E>
__global__ __launch_bounds__(256)
void gemm_f16wmma(const float* __restrict__ A0, const float* __restrict__ A1,
                  const half_t* __restrict__ Wh, const float* __restrict__ bias,
                  float* __restrict__ C,
                  int M, int N, int K, int lda0, int lda1, int ldc, int K0) {
  __shared__ half_t As[BM][BK];
  __shared__ half_t Bs[2][BN][BK];

  const int tid  = threadIdx.x;
  const int lane = tid & 31;
  const int wid  = tid >> 5;
  const int wm   = wid & 3;
  const int wn   = wid >> 2;
  const int m0   = blockIdx.y * BM;
  const int n0   = blockIdx.x * BN;

  const int r16 = lane & 15;
  const int hi  = lane >> 4;
  const int klo = hi * 8;         // A-fragment K sub-offset (ISA layout)
  const int kb  = hi * 8;         // B-fragment K sub-offset (ISA layout)

  // cooperative-load coordinates
  const int arow = tid >> 1;            // 0..127
  const int akc  = (tid & 1) * 16;      // 0 or 16
  const int am   = m0 + arow;
  const int brow = tid >> 2;            // 0..63
  const int bkc  = (tid & 3) * 8;       // 0,8,16,24 (8 halves = 16B)
  const int bgn  = n0 + brow;

  v4f ra[4];        // A chunk (16 f32)
  v4f rz[4];        // silu stream for P==2
  v8h rbh;          // B chunk fallback (8 f16)

  auto loadRegsA = [&](int kt) {
    int kg = kt + akc;
    const float* base;
    if (P == 0) base = A0 + (size_t)am * lda0 + kg;
    else if (P == 1) base = A0 + (size_t)flipRow(am) * lda0 + kg;
    else if (P == 2) {
      base = A0 + (size_t)am * lda0 + kg;
      const float* b2 = A1 + (size_t)am * lda1 + kg;
#pragma unroll
      for (int c = 0; c < 4; ++c) rz[c] = *(const v4f*)(b2 + 4 * c);
    } else {  // P == 3: K0 is BK-aligned, chunk lies entirely on one side
      base = (kg < K0) ? (A0 + (size_t)am * lda0 + kg)
                       : (A1 + (size_t)flipRow(am) * lda1 + (kg - K0));
    }
#pragma unroll
    for (int c = 0; c < 4; ++c) ra[c] = *(const v4f*)(base + 4 * c);
  };

  auto storeA = [&]() {
#pragma unroll
    for (int c = 0; c < 2; ++c) {
      v8h h;
#pragma unroll
      for (int j = 0; j < 8; ++j) {
        float v = ra[c * 2 + (j >> 2)][j & 3];
        if (P == 2) v *= siluf(rz[c * 2 + (j >> 2)][j & 3]);
        h[j] = (half_t)v;
      }
      *(v8h*)&As[arow][akc + c * 8] = h;
    }
  };

  // ---- prologue ----
  if (bgn >= N) {   // ragged rows: zero both buffers once; never touched again
    v8h zh;
#pragma unroll
    for (int j = 0; j < 8; ++j) zh[j] = (half_t)0.f;
    *(v8h*)&Bs[0][brow][bkc] = zh;
    *(v8h*)&Bs[1][brow][bkc] = zh;
  }
#if ASYNC_LDS
  if (bgn < N) asyncCopy16(Wh + (size_t)bgn * K + bkc, &Bs[0][brow][bkc]);
#else
  if (bgn < N) rbh = *(const v8h*)(Wh + (size_t)bgn * K + bkc);
#endif
  loadRegsA(0);
  storeA();
#if ASYNC_LDS
  waitAsync();
#else
  if (bgn < N) *(v8h*)&Bs[0][brow][bkc] = rbh;
#endif
  __syncthreads();

  v8f zero8 = {0.f,0.f,0.f,0.f,0.f,0.f,0.f,0.f};
  v8f acc[2][2] = {{zero8, zero8}, {zero8, zero8}};

  int buf = 0;
  for (int kt = 0; kt < K; kt += BK) {
    const bool more = (kt + BK) < K;
    if (more) {
#if ASYNC_LDS
      if (bgn < N)
        asyncCopy16(Wh + (size_t)bgn * K + (kt + BK) + bkc, &Bs[buf ^ 1][brow][bkc]);
#else
      if (bgn < N) rbh = *(const v8h*)(Wh + (size_t)bgn * K + (kt + BK) + bkc);
#endif
      loadRegsA(kt + BK);          // overlaps the WMMAs below
    }

    // ---- fragments (ISA 16-bit A 16x32 and B 32x16 lane layouts) ----
    v16h af[2], bf[2];
#pragma unroll
    for (int i = 0; i < 2; ++i) {
      int rA = wm * 32 + i * 16 + r16;
      v8h lo8 = *(const v8h*)&As[rA][klo];
      v8h hi8 = *(const v8h*)&As[rA][16 + klo];
#pragma unroll
      for (int j = 0; j < 8; ++j) { af[i][j] = lo8[j]; af[i][8 + j] = hi8[j]; }
    }
#pragma unroll
    for (int i = 0; i < 2; ++i) {
      int rB = wn * 32 + i * 16 + r16;
      v8h lo8 = *(const v8h*)&Bs[buf][rB][kb];
      v8h hi8 = *(const v8h*)&Bs[buf][rB][16 + kb];
#pragma unroll
      for (int j = 0; j < 8; ++j) { bf[i][j] = lo8[j]; bf[i][8 + j] = hi8[j]; }
    }

#pragma unroll
    for (int i = 0; i < 2; ++i)
#pragma unroll
      for (int j = 0; j < 2; ++j)
        acc[i][j] = __builtin_amdgcn_wmma_f32_16x16x32_f16(
            false, af[i], false, bf[j], (short)0, acc[i][j], false, false);

    __syncthreads();               // all reads of As / Bs[buf] complete
    if (more) {
      storeA();
#if !ASYNC_LDS
      if (bgn < N) *(v8h*)&Bs[buf ^ 1][brow][bkc] = rbh;
#endif
    }
#if ASYNC_LDS
    waitAsync();                   // my async writes to Bs[buf^1] landed
#endif
    __syncthreads();
    buf ^= 1;
  }

  // ---- store D (C/D layout: VGPR r -> row r + 8*hi, col = lane&15) ----
#pragma unroll
  for (int i = 0; i < 2; ++i) {
#pragma unroll
    for (int j = 0; j < 2; ++j) {
      int col = n0 + wn * 32 + j * 16 + r16;
      if (col < N) {
        float bv = bias ? bias[col] : 0.f;
#pragma unroll
        for (int r = 0; r < 8; ++r) {
          int row = m0 + wm * 32 + i * 16 + hi * 8 + r;
          C[(size_t)row * ldc + col] = epilog<E>(acc[i][j][r], bv);
        }
      }
    }
  }
}

// f32 -> f16 weight pre-conversion (vector-4)
__global__ __launch_bounds__(256)
void cvt_f32_f16_kernel(const float* __restrict__ s, half_t* __restrict__ d, int n4) {
  int i = blockIdx.x * blockDim.x + threadIdx.x;
  if (i < n4) {
    v4f v = *(const v4f*)(s + 4 * (size_t)i);
    v4h h;
#pragma unroll
    for (int j = 0; j < 4; ++j) h[j] = (half_t)v[j];
    *(v4h*)(d + 4 * (size_t)i) = h;
  }
}

// u = silu(causal depthwise conv(xin) + convb); xin = xz[:, 0:DI], row stride 2*DI
__global__ __launch_bounds__(256)
void conv_silu_kernel(const float* __restrict__ xz, const float* __restrict__ convw,
                      const float* __restrict__ convb, float* __restrict__ u) {
  int idx = blockIdx.x * blockDim.x + threadIdx.x;   // idx = m*DI + d
  if (idx >= MTOK * DI) return;
  int d = idx & (DI - 1);
  int m = idx / DI;
  int t = m & (LSEQ - 1);
  float s = convb[d];
#pragma unroll
  for (int j = 0; j < DCONV; ++j) {
    int tt = t - (DCONV - 1) + j;
    if (tt >= 0) s += convw[d * DCONV + j] * xz[(size_t)(m - (DCONV - 1) + j) * (2 * DI) + d];
  }
  u[idx] = siluf(s);
}

// Selective scan: one thread per (b, d, n); h in a register; 16-lane shuffle reduction;
// 4 timesteps of loads chunked per iteration to hide latency.
__global__ __launch_bounds__(256)
void scan_kernel(const float* __restrict__ u, const float* __restrict__ delta,
                 const float* __restrict__ xdbl, const float* __restrict__ Alog,
                 const float* __restrict__ Dsk, float* __restrict__ y, int ystride) {
  const int tid = threadIdx.x;
  const int n   = tid & 15;
  const int ch  = blockIdx.x * 16 + (tid >> 4);  // 0..B*DI-1
  const int b   = ch / DI;
  const int d   = ch & (DI - 1);

  const float A  = -__expf(Alog[d * DSTATE + n]);
  const float Dv = Dsk[d];
  float h = 0.f;

  for (int t0 = 0; t0 < LSEQ; t0 += 4) {
    float dl[4], ul[4], bl[4], cl[4];
#pragma unroll
    for (int j = 0; j < 4; ++j) {
      size_t row = (size_t)(b * LSEQ + t0 + j);
      dl[j] = delta[row * DI + d];
      ul[j] = u[row * DI + d];
      bl[j] = xdbl[row * XDBL_N + DTRANK + n];
      cl[j] = xdbl[row * XDBL_N + DTRANK + DSTATE + n];
    }
#pragma unroll
    for (int j = 0; j < 4; ++j) {
      float dA = __expf(dl[j] * A);
      h = h * dA + dl[j] * ul[j] * bl[j];
      float p = h * cl[j];
      p += __shfl_xor(p, 8, 16);
      p += __shfl_xor(p, 4, 16);
      p += __shfl_xor(p, 2, 16);
      p += __shfl_xor(p, 1, 16);
      if (n == 0)
        y[(size_t)(b * LSEQ + t0 + j) * ystride + d] = p + ul[j] * Dv;
    }
  }
}

extern "C" void kernel_launch(void* const* d_in, const int* in_sizes, int n_in,
                              void* d_out, int out_size, void* d_ws, size_t ws_size,
                              hipStream_t stream) {
  const float* x = (const float*)d_in[0];
  const float* Pf[11]; const float* Pb[11];
  for (int i = 0; i < 11; ++i) { Pf[i] = (const float*)d_in[1 + i]; Pb[i] = (const float*)d_in[12 + i]; }
  // param order: 0 Win, 1 bin, 2 convw, 3 convb, 4 Wx, 5 Wdt, 6 bdt, 7 Alog, 8 Dsk, 9 Wout, 10 bout
  const float* projW = (const float*)d_in[23];
  const float* projb = (const float*)d_in[24];

  // f32 workspace overlay (~85 MB, reused across directions):
  float* ws   = (float*)d_ws;
  float* xz   = ws;                              // MTOK x 4096 (xin | z); scan y overlays xin half
  float* u    = xz   + (size_t)MTOK * (2 * DI);  // MTOK x DI
  float* xdbl = u    + (size_t)MTOK * DI;        // MTOK x 96
  float* delt = xdbl + (size_t)MTOK * XDBL_N;    // MTOK x DI
  float* outf = delt + (size_t)MTOK * DI;        // MTOK x DMODEL
  float* outb = outf + (size_t)MTOK * DMODEL;    // MTOK x DMODEL
  float* fend = outb + (size_t)MTOK * DMODEL;
  float* outd[2] = {outf, outb};
  // f16 weight staging (~13 MB), reused across directions:
  half_t* whWin  = (half_t*)fend;                       // 4096x1024 (also projW 1024x2048)
  half_t* whWx   = whWin  + (size_t)(2 * DI) * DMODEL;  // 96x2048
  half_t* whWdt  = whWx   + (size_t)XDBL_N * DI;        // 2048x64
  half_t* whWout = whWdt  + (size_t)DI * DTRANK;        // 1024x2048

  dim3 blk(256);
  auto cvt = [&](const float* src, half_t* dst, size_t n) {
    int n4 = (int)(n / 4);
    cvt_f32_f16_kernel<<<dim3((n4 + 255) / 256), blk, 0, stream>>>(src, dst, n4);
  };

  for (int dir = 0; dir < 2; ++dir) {
    const float* const* p = dir ? Pb : Pf;
    // weight pre-conversion for this direction
    cvt(p[0], whWin,  (size_t)(2 * DI) * DMODEL);
    cvt(p[4], whWx,   (size_t)XDBL_N * DI);
    cvt(p[5], whWdt,  (size_t)DI * DTRANK);
    cvt(p[9], whWout, (size_t)DMODEL * DI);
    // 1) xz = x @ Win^T + bin   (backward: time-flipped input)
    dim3 g1((2 * DI) / BN, MTOK / BM);
    if (dir == 0)
      gemm_f16wmma<0,0><<<g1, blk, 0, stream>>>(x, nullptr, whWin, p[1], xz,
          MTOK, 2 * DI, DMODEL, DMODEL, 0, 2 * DI, 0);
    else
      gemm_f16wmma<1,0><<<g1, blk, 0, stream>>>(x, nullptr, whWin, p[1], xz,
          MTOK, 2 * DI, DMODEL, DMODEL, 0, 2 * DI, 0);
    // 2) u = silu(conv(xin) + convb)
    conv_silu_kernel<<<dim3((MTOK * DI) / 256), blk, 0, stream>>>(xz, p[2], p[3], u);
    // 3) xdbl = u @ Wx^T
    dim3 g3((XDBL_N + BN - 1) / BN, MTOK / BM);
    gemm_f16wmma<0,0><<<g3, blk, 0, stream>>>(u, nullptr, whWx, nullptr, xdbl,
        MTOK, XDBL_N, DI, DI, 0, XDBL_N, 0);
    // 4) delta = softplus(xdbl[:, :64] @ Wdt^T + bdt)
    dim3 g4(DI / BN, MTOK / BM);
    gemm_f16wmma<0,1><<<g4, blk, 0, stream>>>(xdbl, nullptr, whWdt, p[6], delt,
        MTOK, DI, DTRANK, XDBL_N, 0, DI, 0);
    // 5) selective scan -> y (overlaid on xin half of xz, stride 2*DI)
    scan_kernel<<<dim3((BATCH_ * DI * DSTATE) / 256), blk, 0, stream>>>(
        u, delt, xdbl, p[7], p[8], xz, 2 * DI);
    // 6) outdir = (y * silu(z)) @ Wout^T + bout
    dim3 g6(DMODEL / BN, MTOK / BM);
    gemm_f16wmma<2,0><<<g6, blk, 0, stream>>>(xz, xz + DI, whWout, p[10], outd[dir],
        MTOK, DMODEL, DI, 2 * DI, 2 * DI, DMODEL, 0);
  }
  // 7) out = concat(out_f, flip(out_b)) @ proj_W^T + proj_b
  cvt(projW, whWin, (size_t)DMODEL * (2 * DMODEL));
  dim3 g7(DMODEL / BN, MTOK / BM);
  gemm_f16wmma<3,0><<<g7, blk, 0, stream>>>(outf, outb, whWin, projb, (float*)d_out,
      MTOK, DMODEL, 2 * DMODEL, DMODEL, DMODEL, DMODEL, DMODEL);
}
